// JMMDLoss_29643864277357
// MI455X (gfx1250) — compile-verified, matrix operand order checked by reading, not run on hardware
//
#include <hip/hip_runtime.h>
#include <hip/hip_bf16.h>
#include <stdint.h>

// ---------------------------------------------------------------------------
// JMMD loss, fully fused for MI455X (gfx1250).
//   N = 8192 total rows (4096 src + 4096 tar), D0 = 2048, D1 = 128.
//   joint[i,j] = (sum_{t=0..4} exp(-l2_0[i,j]/(bw*2^t))) * exp(-l2_1[i,j]/1.68)
//   out = (1/4096^2) * sum_{i,j} s_i s_j joint[i,j],  s_i = +1 (i<4096) / -1
// Strategy: bf16 WMMA (v_wmma_f32_16x16x32_bf16) Gram tiles, upper-triangular
// tile scheduling (symmetry => 2x less work), analytic sum(l2) for bandwidth,
// no 8192^2 intermediate ever touches memory.
// ---------------------------------------------------------------------------

typedef __attribute__((ext_vector_type(16))) __bf16 v16bf;
typedef __attribute__((ext_vector_type(8)))  float  v8f;

#define BHALF   4096
#define NTOT    8192
#define D0      2048
#define D1      128

__device__ __forceinline__ unsigned short f2bf(float x) {
    union { float f; unsigned int u; } c; c.f = x;
    unsigned int u = c.u;
    unsigned int r = (u + 0x7FFFu + ((u >> 16) & 1u)) >> 16;  // RNE
    return (unsigned short)r;
}

union FragU { uint4 q[2]; v16bf v; };

// A-fragment (16x32 bf16) per ISA 7.12.2: lane<16 holds M=lane, K={k..k+7, k+16..k+23};
// lane>=16 holds M=lane-16, K={k+8..k+15, k+24..k+31}. For G = T*T^T the B-fragment
// is loaded with the identical pattern from the column-block's rows of T.
__device__ __forceinline__ v16bf load_frag(const unsigned short* __restrict__ T,
                                           int ld, int row, int k, int sel) {
    const unsigned short* p = T + (size_t)row * ld + k + sel * 8;
    FragU f;
    f.q[0] = *(const uint4*)(p);
    f.q[1] = *(const uint4*)(p + 16);
    return f.v;
}

// ---- kernel 1: per-row sum of squares + bf16 conversion (both feature sets) ----
__global__ void __launch_bounds__(256) prep_kernel(
    const float* __restrict__ s0, const float* __restrict__ s1,
    const float* __restrict__ t0, const float* __restrict__ t1,
    unsigned short* __restrict__ w0, unsigned short* __restrict__ w1,
    float* __restrict__ sq0, float* __restrict__ sq1)
{
    const int r   = blockIdx.x;
    const int tid = threadIdx.x;
    const float* row0 = (r < BHALF) ? (s0 + (size_t)r * D0) : (t0 + (size_t)(r - BHALF) * D0);
    const float* row1 = (r < BHALF) ? (s1 + (size_t)r * D1) : (t1 + (size_t)(r - BHALF) * D1);

    float a0 = 0.0f, a1 = 0.0f;
    for (int c = tid; c < D0; c += 256) {
        float x = row0[c];
        a0 += x * x;
        w0[(size_t)r * D0 + c] = f2bf(x);
    }
    if (tid < D1) {
        float x = row1[tid];
        a1 = x * x;
        w1[(size_t)r * D1 + tid] = f2bf(x);
    }

    __shared__ float sm[256];
    sm[tid] = a0; __syncthreads();
    for (int s = 128; s > 0; s >>= 1) { if (tid < s) sm[tid] += sm[tid + s]; __syncthreads(); }
    if (tid == 0) sq0[r] = sm[0];
    __syncthreads();
    sm[tid] = (tid < D1) ? a1 : 0.0f; __syncthreads();
    for (int s = 128; s > 0; s >>= 1) { if (tid < s) sm[tid] += sm[tid + s]; __syncthreads(); }
    if (tid == 0) sq1[r] = sm[0];
}

// ---- kernel 2: column sums of total_0 (for analytic sum(l2)) ----
__global__ void __launch_bounds__(256) colsum_kernel(
    const float* __restrict__ s0, const float* __restrict__ t0,
    float* __restrict__ colsum)
{
    const int c = blockIdx.x * 256 + threadIdx.x;  // 0..2047
    float acc = 0.0f;
    for (int r = 0; r < BHALF; ++r) acc += s0[(size_t)r * D0 + c];
    for (int r = 0; r < BHALF; ++r) acc += t0[(size_t)r * D0 + c];
    colsum[c] = acc;
}

// ---- kernel 3: bandwidth scalar; zero output accumulator ----
// sum(l2) = 2*N*sum(sq) - 2*||colsum||^2 ; bw = sum(l2)/(N^2-N)/2^(5//2)
__global__ void __launch_bounds__(256) finalize_bw(
    const float* __restrict__ sq0, const float* __restrict__ colsum,
    float* __restrict__ scalars, float* __restrict__ out)
{
    __shared__ float sm[256];
    const int tid = threadIdx.x;

    float s = 0.0f;
    for (int i = tid; i < NTOT; i += 256) s += sq0[i];
    sm[tid] = s; __syncthreads();
    for (int k = 128; k > 0; k >>= 1) { if (tid < k) sm[tid] += sm[tid + k]; __syncthreads(); }
    float S = sm[0]; __syncthreads();

    float cc = 0.0f;
    for (int i = tid; i < D0; i += 256) { float v = colsum[i]; cc += v * v; }
    sm[tid] = cc; __syncthreads();
    for (int k = 128; k > 0; k >>= 1) { if (tid < k) sm[tid] += sm[tid + k]; __syncthreads(); }

    if (tid == 0) {
        double N = (double)NTOT;
        double suml2 = 2.0 * N * (double)S - 2.0 * (double)sm[0];
        double bw = suml2 / (N * N - N) / 4.0;   // / kernel_mul^(kernel_num//2) = 2^2
        scalars[0] = (float)(1.0 / bw);          // inverse base bandwidth
        out[0] = 0.0f;                           // zero the scalar accumulator
    }
}

// ---- kernel 4: fused Gram + RBF + signed reduction over upper-triangular tiles ----
__global__ void __launch_bounds__(256) jmmd_main(
    const unsigned short* __restrict__ T0, const unsigned short* __restrict__ T1,
    const float* __restrict__ sq0, const float* __restrict__ sq1,
    const float* __restrict__ scalars, float* __restrict__ out)
{
    const int bi = blockIdx.y, bj = blockIdx.x;
    if (bj < bi) return;  // symmetry: upper triangle only (uniform per block)

    __shared__ float sqr0[128], sqc0[128], sqr1[128], sqc1[128];
    __shared__ float k1buf[128 * 128];   // exp(-l2_1/1.68) tile (frees VGPRs for big loop)
    __shared__ float red[256];

    const int tid = threadIdx.x;
    if (tid < 128) {
        sqr0[tid] = sq0[bi * 128 + tid];
        sqc0[tid] = sq0[bj * 128 + tid];
        sqr1[tid] = sq1[bi * 128 + tid];
        sqc1[tid] = sq1[bj * 128 + tid];
    }
    __syncthreads();

    const int lane = tid & 31;
    const int w    = tid >> 5;     // 8 waves
    const int wm   = w >> 1;       // 0..3 -> 32-row strip
    const int wn   = w & 1;        // 0..1 -> 64-col strip
    const int mlan = lane & 15;
    const int sel  = lane >> 4;

    const int rowb = bi * 128 + wm * 32;
    const int colb = bj * 128 + wn * 64;

    const v8f vzero = {0.f, 0.f, 0.f, 0.f, 0.f, 0.f, 0.f, 0.f};

    // ---------- phase 1: f1 Gram (K = 128) -> k1 tile in LDS ----------
    v8f acc1[2][4];
#pragma unroll
    for (int a = 0; a < 2; ++a)
#pragma unroll
        for (int b = 0; b < 4; ++b) acc1[a][b] = vzero;

#pragma unroll
    for (int k = 0; k < D1; k += 32) {
        v16bf amat[2], bmat[4];
#pragma unroll
        for (int a = 0; a < 2; ++a) amat[a] = load_frag(T1, D1, rowb + a * 16 + mlan, k, sel);
#pragma unroll
        for (int b = 0; b < 4; ++b) bmat[b] = load_frag(T1, D1, colb + b * 16 + mlan, k, sel);
#pragma unroll
        for (int a = 0; a < 2; ++a)
#pragma unroll
            for (int b = 0; b < 4; ++b)
                acc1[a][b] = __builtin_amdgcn_wmma_f32_16x16x32_bf16(
                    false, amat[a], false, bmat[b], (short)0, acc1[a][b], false, false);
    }

    const float inv_s1 = 1.0f / 1.68f;
#pragma unroll
    for (int a = 0; a < 2; ++a)
#pragma unroll
        for (int b = 0; b < 4; ++b)
#pragma unroll
            for (int r = 0; r < 8; ++r) {
                int il = wm * 32 + a * 16 + r + sel * 8;   // C layout: M = r + 8*(lane>=16)
                int jl = wn * 64 + b * 16 + mlan;          // N = lane&15
                float l2 = sqr1[il] + sqc1[jl] - 2.0f * acc1[a][b][r];
                k1buf[il * 128 + jl] = __expf(-l2 * inv_s1);
            }

    // ---------- phase 2: f0 Gram (K = 2048) ----------
    v8f acc0[2][4];
#pragma unroll
    for (int a = 0; a < 2; ++a)
#pragma unroll
        for (int b = 0; b < 4; ++b) acc0[a][b] = vzero;

    for (int k = 0; k < D0; k += 32) {
        v16bf amat[2], bmat[4];
#pragma unroll
        for (int a = 0; a < 2; ++a) amat[a] = load_frag(T0, D0, rowb + a * 16 + mlan, k, sel);
#pragma unroll
        for (int b = 0; b < 4; ++b) bmat[b] = load_frag(T0, D0, colb + b * 16 + mlan, k, sel);
#pragma unroll
        for (int a = 0; a < 2; ++a)
#pragma unroll
            for (int b = 0; b < 4; ++b)
                acc0[a][b] = __builtin_amdgcn_wmma_f32_16x16x32_bf16(
                    false, amat[a], false, bmat[b], (short)0, acc0[a][b], false, false);
    }

    // ---------- epilogue: 5-bandwidth RBF, joint product, signed/weighted sum ----------
    const float ib0 = scalars[0];
    float local = 0.0f;
#pragma unroll
    for (int a = 0; a < 2; ++a)
#pragma unroll
        for (int b = 0; b < 4; ++b)
#pragma unroll
            for (int r = 0; r < 8; ++r) {
                int il = wm * 32 + a * 16 + r + sel * 8;
                int jl = wn * 64 + b * 16 + mlan;
                float l2 = sqr0[il] + sqc0[jl] - 2.0f * acc0[a][b][r];
                float e = 0.0f, sc = ib0;
#pragma unroll
                for (int t = 0; t < 5; ++t) { e += __expf(-l2 * sc); sc *= 0.5f; }
                float joint = e * k1buf[il * 128 + jl];
                int ig = bi * 128 + il, jg = bj * 128 + jl;
                float wgt = (bi != bj) ? 2.0f
                                       : (jg > ig ? 2.0f : (jg == ig ? 1.0f : 0.0f));
                float sgn = ((ig < BHALF) == (jg < BHALF)) ? 1.0f : -1.0f;
                local += wgt * sgn * joint;
            }

    red[tid] = local; __syncthreads();
    for (int s = 128; s > 0; s >>= 1) { if (tid < s) red[tid] += red[tid + s]; __syncthreads(); }
    if (tid == 0)
        atomicAdd(out, red[0] * (1.0f / ((float)BHALF * (float)BHALF)));
}

extern "C" void kernel_launch(void* const* d_in, const int* in_sizes, int n_in,
                              void* d_out, int out_size, void* d_ws, size_t ws_size,
                              hipStream_t stream) {
    const float* src_f0 = (const float*)d_in[0];
    const float* src_f1 = (const float*)d_in[1];
    const float* tar_f0 = (const float*)d_in[2];
    const float* tar_f1 = (const float*)d_in[3];
    float* out = (float*)d_out;

    // workspace carve (~35.7 MB): bf16 copies + sq vectors + colsum + scalars
    char* p = (char*)d_ws;
    unsigned short* w0 = (unsigned short*)p; p += (size_t)NTOT * D0 * 2;
    unsigned short* w1 = (unsigned short*)p; p += (size_t)NTOT * D1 * 2;
    float* sq0     = (float*)p; p += (size_t)NTOT * 4;
    float* sq1     = (float*)p; p += (size_t)NTOT * 4;
    float* colsum  = (float*)p; p += (size_t)D0 * 4;
    float* scalars = (float*)p; p += 256;

    prep_kernel<<<NTOT, 256, 0, stream>>>(src_f0, src_f1, tar_f0, tar_f1, w0, w1, sq0, sq1);
    colsum_kernel<<<D0 / 256, 256, 0, stream>>>(src_f0, tar_f0, colsum);
    finalize_bw<<<1, 256, 0, stream>>>(sq0, colsum, scalars, out);

    dim3 grid(NTOT / 128, NTOT / 128);  // 64 x 64, lower triangle exits immediately
    jmmd_main<<<grid, 256, 0, stream>>>(w0, w1, sq0, sq1, scalars, out);
}
